// OptimizedDiffusionFusion_72945724555744
// MI455X (gfx1250) — matrix operand order repaired and checked by reading, not compiled, not run on hardware
//
#include <hip/hip_runtime.h>

#define B_    4
#define CIN_  64
#define H_    192
#define W_    192
#define HW_   (H_ * W_)
#define PT_   (HW_ / 16)     // 2304 pixel tiles per image
#define EDGE_ 36
#define MID_  64
#define KSEL_ 32
#define WPB_  8              // waves per block (256 threads)

typedef __attribute__((ext_vector_type(16))) __bf16 v16bf;
typedef __attribute__((ext_vector_type(8)))  __bf16 v8bf;
typedef __attribute__((ext_vector_type(8)))  float  v8f;

static __device__ inline v8f wmma_bf16(v16bf a, v16bf b, v8f c) {
  return __builtin_amdgcn_wmma_f32_16x16x32_bf16(false, a, false, b, (short)0, c,
                                                 false, false);
}

// A fragment from bf16 pixel-major tensor baseT[pixel][S]: two aligned b128 loads.
static __device__ inline v16bf load_A_pm(const __bf16* __restrict__ baseT, int S,
                                         int p0, int kofs) {
  const int lane = threadIdx.x & 31;
  const int m    = lane & 15;
  const int kbA  = (lane < 16) ? 0 : 8;
  const __bf16* row = baseT + (size_t)(p0 + m) * S + kofs + kbA;
  const v8bf lo = *(const v8bf*)(row);
  const v8bf hi = *(const v8bf*)(row + 16);
  v16bf a;
#pragma unroll
  for (int j = 0; j < 8; ++j) { a[j] = lo[j]; a[j + 8] = hi[j]; }
  return a;
}

// B fragment from bf16 row-major weights W[o][S]: two aligned b128 loads.
static __device__ inline v16bf load_B_rm(const __bf16* __restrict__ W, int S,
                                         int o, int kofs) {
  const int lane = threadIdx.x & 31;
  const int kb   = (lane < 16) ? 0 : 8;
  const __bf16* row = W + (size_t)o * S + kofs + kb;
  const v8bf lo = *(const v8bf*)(row);
  const v8bf hi = *(const v8bf*)(row + 16);
  v16bf b;
#pragma unroll
  for (int j = 0; j < 8; ++j) { b[j] = lo[j]; b[j + 8] = hi[j]; }
  return b;
}

// ---------- Prepass A: weights -> bf16 padded layouts ----------
__global__ void k_prep_w(const float* __restrict__ We, const float* __restrict__ Wcoo,
                         const float* __restrict__ bcoo, const float* __restrict__ Wpd,
                         const float* __restrict__ Wpr, const float* __restrict__ Wprd,
                         __bf16* __restrict__ wE, __bf16* __restrict__ wC,
                         float* __restrict__ bcP, __bf16* __restrict__ wPd,
                         __bf16* __restrict__ wPr, __bf16* __restrict__ wP) {
  int i = blockIdx.x * blockDim.x + threadIdx.x;
  if (i < 36864) {                             // wE[tap][o][c] = We[o][c][tap]
    const int tap = i >> 12, r = i & 4095, o = r >> 6, c = r & 63;
    wE[i] = (__bf16)We[(size_t)(o * 64 + c) * 9 + tap];
    return;
  }
  i -= 36864;
  if (i < 4096) {                              // wC: 64 rows, zero-pad o>=36
    const int o = i >> 6, c = i & 63;
    wC[i] = (__bf16)((o < EDGE_) ? Wcoo[o * 64 + c] : 0.f);
    return;
  }
  i -= 4096;
  if (i < 64) { bcP[i] = (i < EDGE_) ? bcoo[i] : 0.f; return; }
  i -= 64;
  if (i < 4096) { wPd[i] = (__bf16)Wpd[i]; return; }
  i -= 4096;
  if (i < 4096) { wPr[i] = (__bf16)Wpr[i]; return; }
  i -= 4096;
  if (i < 704 * 64) {                          // wP[row][c], row<640 & c<36 valid
    const int row = i >> 6, c = i & 63;
    wP[i] = (__bf16)((row < 640 && c < EDGE_) ? Wprd[row * EDGE_ + c] : 0.f);
  }
}

// ---------- Prepass B: channel-major fp32 -> pixel-major bf16 (LDS transpose) ----------
__global__ void k_transpose(const float* __restrict__ src, __bf16* __restrict__ dst) {
  __shared__ float tile[64][65];
  const int blk = blockIdx.x;
  const int p0  = (blk % (HW_ / 64)) * 64;
  const int b   = blk / (HW_ / 64);
  const float* s = src + (size_t)b * 64 * HW_;
  __bf16* d = dst + ((size_t)b * HW_ + p0) * 64;
#pragma unroll
  for (int i = 0; i < 16; ++i) {               // coalesced along pixels
    const int idx = i * 256 + threadIdx.x;
    const int c = idx >> 6, p = idx & 63;
    tile[p][c] = s[(size_t)c * HW_ + p0 + p];
  }
  __syncthreads();
#pragma unroll
  for (int i = 0; i < 16; ++i) {               // contiguous bf16 along channels
    const int idx = i * 256 + threadIdx.x;
    const int p = idx >> 6, c = idx & 63;
    d[(size_t)p * 64 + c] = (__bf16)tile[p][c];
  }
}

// ---------- Pass 1: 3x3 conv (edge): 16 pixels x ALL 64 out-ch per wave ----------
__global__ void k_edge_conv(const __bf16* __restrict__ t1, const __bf16* __restrict__ wE,
                            const float* __restrict__ be, __bf16* __restrict__ edgeT) {
  const int g  = blockIdx.x * WPB_ + (threadIdx.x >> 5);
  const int pt = g % PT_;
  const int b  = g / PT_;
  const int p0 = pt * 16;
  const int y0 = p0 / W_, x0 = p0 % W_;        // W_%16==0: tile stays in one row
  const int lane = threadIdx.x & 31;
  const int m    = lane & 15;
  const int kbA  = (lane < 16) ? 0 : 8;
  const __bf16* tb = t1 + (size_t)b * HW_ * 64;
  v8f acc[4] = {{}, {}, {}, {}};
  for (int tap = 0; tap < 9; ++tap) {
    const int dy = tap / 3 - 1, dx = tap % 3 - 1;
    const int yy = y0 + dy;
    if (yy < 0 || yy >= H_) continue;          // uniform skip: zero-pad rows
    const int  xx  = x0 + m + dx;
    const bool xv  = (xx >= 0) && (xx < W_);
    const int  xxc = xx < 0 ? 0 : (xx > W_ - 1 ? W_ - 1 : xx);
    const __bf16* prow = tb + ((size_t)yy * W_ + xxc) * 64 + kbA;
#pragma unroll
    for (int ks = 0; ks < 2; ++ks) {           // c = 0..31, 32..63
      v8bf lo = *(const v8bf*)(prow + ks * 32);
      v8bf hi = *(const v8bf*)(prow + ks * 32 + 16);
      if (!xv) {                               // cndmask zeroing, no branchy loads
#pragma unroll
        for (int j = 0; j < 8; ++j) { lo[j] = (__bf16)0.f; hi[j] = (__bf16)0.f; }
      }
      v16bf a;
#pragma unroll
      for (int j = 0; j < 8; ++j) { a[j] = lo[j]; a[j + 8] = hi[j]; }
#pragma unroll
      for (int ot = 0; ot < 4; ++ot) {         // A reused across 4 N-tiles
        const int o = ot * 16 + (lane & 15);
        acc[ot] = wmma_bf16(a, load_B_rm(wE, 64, tap * 64 + o, ks * 32), acc[ot]);
      }
    }
  }
  const int ro = (lane < 16) ? 0 : 8;
#pragma unroll
  for (int ot = 0; ot < 4; ++ot) {
    const int o = ot * 16 + (lane & 15);
    const float bias = be[o];
    __bf16* ob = edgeT + ((size_t)b * HW_ + p0 + ro) * 64 + o;
#pragma unroll
    for (int r = 0; r < 8; ++r) ob[(size_t)r * 64] = (__bf16)(acc[ot][r] + bias);
  }
}

// ---------- Pass 2: 1x1 conv -> edge_prior f32 + bf16 pixel-major copy ----------
__global__ void k_prior(const __bf16* __restrict__ edgeT, const __bf16* __restrict__ wC,
                        const float* __restrict__ bcP, float* __restrict__ eprior,
                        __bf16* __restrict__ epT) {
  const int g  = blockIdx.x * WPB_ + (threadIdx.x >> 5);
  const int pt = g % PT_;
  const int b  = g / PT_;
  const int p0 = pt * 16;
  const int lane = threadIdx.x & 31;
  const __bf16* inb = edgeT + (size_t)b * HW_ * 64;
  v8f acc[4] = {{}, {}, {}, {}};
#pragma unroll
  for (int ks = 0; ks < 2; ++ks) {
    const v16bf a = load_A_pm(inb, 64, p0, ks * 32);
#pragma unroll
    for (int ot = 0; ot < 4; ++ot) {
      const int o = ot * 16 + (lane & 15);     // rows >=36 zero-padded
      acc[ot] = wmma_bf16(a, load_B_rm(wC, 64, o, ks * 32), acc[ot]);
    }
  }
  const int ro = (lane < 16) ? 0 : 8;
#pragma unroll
  for (int ot = 0; ot < 4; ++ot) {
    const int o = ot * 16 + (lane & 15);
    const float bias = bcP[o];
    __bf16* tb = epT + ((size_t)b * HW_ + p0 + ro) * 64 + o;
    float* fb  = eprior + ((size_t)b * EDGE_ + o) * HW_ + p0 + ro;
#pragma unroll
    for (int r = 0; r < 8; ++r) {
      const float v = acc[ot][r] + bias;
      tb[(size_t)r * 64] = (__bf16)v;          // pad channels get exact zeros
      if (o < EDGE_) fb[r] = v;
    }
  }
}

// ---------- Pass 3: spatial mean of edge_prior ----------
__global__ void k_mean(const float* __restrict__ eprior, float* __restrict__ meanb) {
  const int b = blockIdx.x / EDGE_;
  const int e = blockIdx.x % EDGE_;
  const float* src = eprior + ((size_t)b * EDGE_ + e) * HW_;
  float s = 0.f;
  for (int i = threadIdx.x; i < HW_; i += blockDim.x) s += src[i];
  __shared__ float red[256];
  red[threadIdx.x] = s; __syncthreads();
  for (int st = 128; st > 0; st >>= 1) {
    if ((int)threadIdx.x < st) red[threadIdx.x] += red[threadIdx.x + st];
    __syncthreads();
  }
  if (threadIdx.x == 0) meanb[b * EDGE_ + e] = red[0] / (float)HW_;
}

// ---------- Pass 4: scores (linear through mean) + deterministic top-32 ----------
__global__ void k_topk(const float* __restrict__ meanb, const float* __restrict__ Wp,
                       const float* __restrict__ bp, int* __restrict__ idxb) {
  const int b = blockIdx.x;
  __shared__ float sc[MID_];
  const int c = threadIdx.x;
  if (c < MID_) {
    float s = bp[c];
    for (int e = 0; e < EDGE_; ++e) s += Wp[(size_t)c * EDGE_ + e] * meanb[b * EDGE_ + e];
    sc[c] = s;                                 // sigmoid monotonic -> same top-k
  }
  __syncthreads();
  if (threadIdx.x == 0) {
    unsigned long long used = 0ull;
    for (int k = 0; k < KSEL_; ++k) {
      int best = 0; float bv = -1e30f;
      for (int q = 0; q < MID_; ++q)
        if (!((used >> q) & 1ull) && sc[q] > bv) { bv = sc[q]; best = q; }
      used |= (1ull << best);
      idxb[b * KSEL_ + k] = best;
    }
  }
}

// ---------- Prepass C (post-topk): gather W_out columns -> wG[b][o][k] bf16 ----------
__global__ void k_gatherW(const float* __restrict__ Wout, const int* __restrict__ idxb,
                          __bf16* __restrict__ wG) {
  const int i = blockIdx.x * blockDim.x + threadIdx.x;
  if (i >= B_ * 64 * KSEL_) return;
  const int k = i & 31, r = i >> 5, o = r & 63, b = r >> 6;
  wG[i] = (__bf16)Wout[o * MID_ + idxb[b * KSEL_ + k]];
}

// ---------- Pass 5: active_feat = relu(featd * featr): both 16-ch tiles per wave ----------
__global__ void k_feat(const __bf16* __restrict__ t1, const __bf16* __restrict__ t2,
                       const __bf16* __restrict__ wPd, const float* __restrict__ bpd,
                       const __bf16* __restrict__ wPr, const float* __restrict__ bpr,
                       const int* __restrict__ idxb, float* __restrict__ x0) {
  const int g  = blockIdx.x * WPB_ + (threadIdx.x >> 5);
  const int pt = g % PT_;
  const int b  = g / PT_;
  const int p0 = pt * 16;
  const int lane = threadIdx.x & 31;
  const int ch0 = idxb[b * KSEL_ + (lane & 15)];
  const int ch1 = idxb[b * KSEL_ + 16 + (lane & 15)];
  const __bf16* t1b = t1 + (size_t)b * HW_ * 64;
  const __bf16* t2b = t2 + (size_t)b * HW_ * 64;
  v8f ad0 = {}, ad1 = {}, ar0 = {}, ar1 = {};
#pragma unroll
  for (int ks = 0; ks < 2; ++ks) {
    const v16bf a1 = load_A_pm(t1b, 64, p0, ks * 32);
    ad0 = wmma_bf16(a1, load_B_rm(wPd, 64, ch0, ks * 32), ad0);
    ad1 = wmma_bf16(a1, load_B_rm(wPd, 64, ch1, ks * 32), ad1);
    const v16bf a2 = load_A_pm(t2b, 64, p0, ks * 32);
    ar0 = wmma_bf16(a2, load_B_rm(wPr, 64, ch0, ks * 32), ar0);
    ar1 = wmma_bf16(a2, load_B_rm(wPr, 64, ch1, ks * 32), ar1);
  }
  const int ro = (lane < 16) ? 0 : 8;
  float* ob0 = x0 + ((size_t)b * KSEL_ + (lane & 15)) * HW_ + p0 + ro;
  float* ob1 = ob0 + (size_t)16 * HW_;
  const float bd0 = bpd[ch0], br0 = bpr[ch0], bd1 = bpd[ch1], br1 = bpr[ch1];
#pragma unroll
  for (int r = 0; r < 8; ++r) {
    const float v0 = (ad0[r] + bd0) * (ar0[r] + br0);
    const float v1 = (ad1[r] + bd1) * (ar1[r] + br1);
    ob0[r] = v0 > 0.f ? v0 : 0.f;
    ob1[r] = v1 > 0.f ? v1 : 0.f;
  }
}

// ---------- Pass 6: dynamic 9-tap kernels + softmax, stored bf16 ----------
__global__ void k_dynk(const __bf16* __restrict__ epT, const __bf16* __restrict__ wP,
                       const float* __restrict__ bp, const int* __restrict__ idxb,
                       __bf16* __restrict__ kern) {
  const int g  = blockIdx.x * WPB_ + (threadIdx.x >> 5);
  const int wv = threadIdx.x >> 5;
  const int pt = g % PT_;
  const int rest = g / PT_;
  const int s  = rest % KSEL_;
  const int b  = rest / KSEL_;
  const int ch = idxb[b * KSEL_ + s];
  const int p0 = pt * 16;
  const int lane = threadIdx.x & 31;
  const int t   = lane & 15;
  const int row = MID_ + ch * 9 + t;           // wP padded to 704 zero rows
  const __bf16* epb = epT + (size_t)b * HW_ * 64;
  v8f acc = {};
#pragma unroll
  for (int ks = 0; ks < 2; ++ks) {
    acc = wmma_bf16(load_A_pm(epb, 64, p0, ks * 32), load_B_rm(wP, 64, row, ks * 32),
                    acc);
  }
  __shared__ float tile[WPB_][16][17];
  const float bias = bp[row < 640 ? row : 0];  // only t<9 columns consumed
  const int ro = (lane < 16) ? 0 : 8;
#pragma unroll
  for (int r = 0; r < 8; ++r) tile[wv][r + ro][t] = acc[r] + bias;
  __syncthreads();
  if (lane < 16) {
    const int m = lane;
    float mx = tile[wv][m][0];
#pragma unroll
    for (int q = 1; q < 9; ++q) mx = fmaxf(mx, tile[wv][m][q]);
    float e[9], sum = 0.f;
#pragma unroll
    for (int q = 0; q < 9; ++q) { e[q] = __expf(tile[wv][m][q] - mx); sum += e[q]; }
    const float inv = 1.f / sum;
    const size_t base = (((size_t)b * KSEL_ + s) * 9) * HW_ + p0 + m;
#pragma unroll
    for (int q = 0; q < 9; ++q) kern[base + (size_t)q * HW_] = (__bf16)(e[q] * inv);
  }
}

// ---------- Pass 7a: reflect-padded dynamic 3x3 filtering (f32 -> f32) ----------
__global__ void k_filter(const float* __restrict__ xin, const __bf16* __restrict__ kern,
                         float* __restrict__ xout) {
  const int n = blockIdx.x * blockDim.x + threadIdx.x;
  if (n >= B_ * KSEL_ * HW_) return;
  const int p  = n % HW_;
  const int bs = n / HW_;
  const int y = p / W_, x = p % W_;
  const float* src = xin + (size_t)bs * HW_;
  const __bf16* kb = kern + (size_t)bs * 9 * HW_ + p;
  float acc = 0.f;
#pragma unroll
  for (int t = 0; t < 9; ++t) {
    const int dy = t / 3 - 1, dx = t % 3 - 1;
    int yy = y + dy; yy = yy < 0 ? -yy : (yy >= H_ ? 2 * H_ - 2 - yy : yy);
    int xx = x + dx; xx = xx < 0 ? -xx : (xx >= W_ ? 2 * W_ - 2 - xx : xx);
    acc += (float)kb[(size_t)t * HW_] * src[yy * W_ + xx];
  }
  xout[n] = acc;
}

// ---------- Pass 7b: second filter step, writes pixel-major bf16 for out-GEMM ----------
__global__ void k_filter2(const float* __restrict__ xin, const __bf16* __restrict__ kern,
                          __bf16* __restrict__ xfT) {
  const int n = blockIdx.x * blockDim.x + threadIdx.x;
  if (n >= B_ * KSEL_ * HW_) return;
  const int p  = n % HW_;
  const int bs = n / HW_;
  const int s = bs & 31, b = bs >> 5;
  const int y = p / W_, x = p % W_;
  const float* src = xin + (size_t)bs * HW_;
  const __bf16* kb = kern + (size_t)bs * 9 * HW_ + p;
  float acc = 0.f;
#pragma unroll
  for (int t = 0; t < 9; ++t) {
    const int dy = t / 3 - 1, dx = t % 3 - 1;
    int yy = y + dy; yy = yy < 0 ? -yy : (yy >= H_ ? 2 * H_ - 2 - yy : yy);
    int xx = x + dx; xx = xx < 0 ? -xx : (xx >= W_ ? 2 * W_ - 2 - xx : xx);
    acc += (float)kb[(size_t)t * HW_] * src[yy * W_ + xx];
  }
  xfT[((size_t)b * HW_ + p) * KSEL_ + s] = (__bf16)acc;
}

// ---------- Pass 8: gathered 1x1 out-proj + BN + ReLU: 4 N-tiles per wave ----------
__global__ void k_out(const __bf16* __restrict__ xfT, const __bf16* __restrict__ wG,
                      const float* __restrict__ bo, const float* __restrict__ gam,
                      const float* __restrict__ bet, const float* __restrict__ mu,
                      const float* __restrict__ var, float* __restrict__ out) {
  const int g  = blockIdx.x * WPB_ + (threadIdx.x >> 5);
  const int pt = g % PT_;
  const int b  = g / PT_;
  const int p0 = pt * 16;
  const int lane = threadIdx.x & 31;
  const v16bf a = load_A_pm(xfT + (size_t)b * HW_ * KSEL_, KSEL_, p0, 0);
  const __bf16* wGb = wG + (size_t)b * 64 * KSEL_;
  v8f acc[4] = {{}, {}, {}, {}};
#pragma unroll
  for (int ot = 0; ot < 4; ++ot) {
    const int o = ot * 16 + (lane & 15);
    acc[ot] = wmma_bf16(a, load_B_rm(wGb, KSEL_, o, 0), acc[ot]);
  }
  const int ro = (lane < 16) ? 0 : 8;
#pragma unroll
  for (int ot = 0; ot < 4; ++ot) {
    const int o = ot * 16 + (lane & 15);
    const float scale = gam[o] * rsqrtf(var[o] + 1e-5f);
    const float shift = bet[o] - mu[o] * scale;
    const float bias  = bo[o];
    float* outb = out + ((size_t)b * MID_ + o) * HW_ + p0 + ro;
#pragma unroll
    for (int r = 0; r < 8; ++r) {
      const float v = (acc[ot][r] + bias) * scale + shift;
      outb[r] = v > 0.f ? v : 0.f;
    }
  }
}

extern "C" void kernel_launch(void* const* d_in, const int* in_sizes, int n_in,
                              void* d_out, int out_size, void* d_ws, size_t ws_size,
                              hipStream_t stream) {
  (void)in_sizes; (void)n_in; (void)out_size; (void)ws_size;
  const float* g1   = (const float*)d_in[0];
  const float* g2   = (const float*)d_in[1];
  const float* We   = (const float*)d_in[2];
  const float* be   = (const float*)d_in[3];
  const float* Wcoo = (const float*)d_in[4];
  const float* bcoo = (const float*)d_in[5];
  const float* Wprd = (const float*)d_in[6];
  const float* bprd = (const float*)d_in[7];
  const float* Wpd  = (const float*)d_in[8];
  const float* bpd  = (const float*)d_in[9];
  const float* Wpr  = (const float*)d_in[10];
  const float* bpr  = (const float*)d_in[11];
  const float* Wout = (const float*)d_in[12];
  const float* bout = (const float*)d_in[13];
  const float* gam  = (const float*)d_in[14];
  const float* bet  = (const float*)d_in[15];
  const float* mu   = (const float*)d_in[16];
  const float* var  = (const float*)d_in[17];

  float* out    = (float*)d_out;                     // (B,64,H,W)
  float* eprior = out + (size_t)B_ * MID_ * HW_;     // (B,36,H,W)

  char* w = (char*)d_ws;                             // ~160.7 MB, with aliasing
  __bf16* t1    = (__bf16*)(w);                      // 18,874,368 B  (later: x1 f32)
  __bf16* t2    = (__bf16*)(w + 18874368u);          // 18,874,368 B  (later: xfT bf16)
  __bf16* edgeT = (__bf16*)(w + 37748736u);          // 18,874,368 B  (later: x0 f32)
  __bf16* epT   = (__bf16*)(w + 56623104u);          // 18,874,368 B
  __bf16* kern  = (__bf16*)(w + 75497472u);          // 84,934,656 B
  __bf16* wE    = (__bf16*)(w + 160432128u);         //     73,728 B
  __bf16* wC    = (__bf16*)(w + 160505856u);         //      8,192 B
  __bf16* wPdB  = (__bf16*)(w + 160514048u);         //      8,192 B
  __bf16* wPrB  = (__bf16*)(w + 160522240u);         //      8,192 B
  __bf16* wP    = (__bf16*)(w + 160530432u);         //     90,112 B (704x64)
  __bf16* wG    = (__bf16*)(w + 160620544u);         //     16,384 B
  float*  bcP   = (float*)(w + 160636928u);          //        256 B
  float*  meanb = (float*)(w + 160637184u);          //        576 B
  int*    idxb  = (int*)(w + 160637760u);            //        512 B
  float*  x0    = (float*)edgeT;                     // alias (edgeT dead after k_prior)
  float*  x1    = (float*)t1;                        // alias (t1 dead after k_feat)
  __bf16* xfT   = (__bf16*)t2;                       // alias (t2 dead after k_feat)

  k_prep_w   <<<(94272 + 255) / 256, 256, 0, stream>>>(We, Wcoo, bcoo, Wpd, Wpr, Wprd,
                                                       wE, wC, bcP, wPdB, wPrB, wP);
  k_transpose<<<B_ * (HW_ / 64), 256, 0, stream>>>(g1, t1);
  k_transpose<<<B_ * (HW_ / 64), 256, 0, stream>>>(g2, t2);
  k_edge_conv<<<B_ * PT_ / WPB_, 32 * WPB_, 0, stream>>>(t1, wE, be, edgeT);
  k_prior    <<<B_ * PT_ / WPB_, 32 * WPB_, 0, stream>>>(edgeT, wC, bcP, eprior, epT);
  k_mean     <<<B_ * EDGE_, 256, 0, stream>>>(eprior, meanb);
  k_topk     <<<B_, 64, 0, stream>>>(meanb, Wprd, bprd, idxb);
  k_gatherW  <<<(B_ * 64 * KSEL_ + 255) / 256, 256, 0, stream>>>(Wout, idxb, wG);
  k_feat     <<<B_ * PT_ / WPB_, 32 * WPB_, 0, stream>>>(t1, t2, wPdB, bpd, wPrB, bpr,
                                                         idxb, x0);
  k_dynk     <<<B_ * KSEL_ * PT_ / WPB_, 32 * WPB_, 0, stream>>>(epT, wP, bprd, idxb,
                                                                 kern);
  const int nflt = B_ * KSEL_ * HW_;
  k_filter   <<<(nflt + 255) / 256, 256, 0, stream>>>(x0, kern, x1);    // step 1
  k_filter2  <<<(nflt + 255) / 256, 256, 0, stream>>>(x1, kern, xfT);   // step 2
  k_out      <<<B_ * PT_ / WPB_, 32 * WPB_, 0, stream>>>(xfT, wG, bout, gam, bet, mu,
                                                         var, out);
}
